// Block_4922032521392
// MI455X (gfx1250) — compile-verified, matrix operand order checked by reading, not run on hardware
//
#include <hip/hip_runtime.h>

typedef __attribute__((ext_vector_type(16))) __bf16 v16bf;
typedef __attribute__((ext_vector_type(8)))  float  v8f;

#define DEV __device__ __forceinline__

DEV v8f wmma_bf16(v16bf a, v16bf b, v8f c) {
  return __builtin_amdgcn_wmma_f32_16x16x32_bf16(
      /*neg_a=*/false, a, /*neg_b=*/false, b,
      /*c_mod=*/(short)0, c, /*reuse_a=*/false, /*reuse_b=*/false);
}

// ---- Fragment loaders (wave32, 16x16x32 bf16 WMMA) ----------------------
// A / W-row fragment (16 rows x 32 K, row-major, leading dim ld):
// lane L holds row L&15; VGPR r holds K pair k = 8*(L>=16) + 16*(r>>2) + 2*(r&3).
// r=0..3 and r=4..7 are each 16B contiguous -> vectorizes to global_load_b128.
DEV v16bf frag_row_bf16(const __bf16* p, int ld) {
  const int lane = threadIdx.x & 31;
  const int row  = lane & 15;
  const int kb   = (lane >> 4) << 3;
  union { v16bf v; unsigned u[8]; } f;
#pragma unroll
  for (int r = 0; r < 8; ++r) {
    const int k = kb + ((r >> 2) << 4) + ((r & 3) << 1);
    f.u[r] = *reinterpret_cast<const unsigned*>(p + row * ld + k);
  }
  return f.v;
}

// B fragment from a [K x N] row-major (non-transposed) source: lane = column,
// K pairs strided by ld -> two b16 loads packed per VGPR.
DEV v16bf frag_col_bf16(const __bf16* p, int ld) {
  const int lane = threadIdx.x & 31;
  const int col  = lane & 15;
  const int kb   = (lane >> 4) << 3;
  union { v16bf v; unsigned u[8]; } f;
#pragma unroll
  for (int r = 0; r < 8; ++r) {
    const int k = kb + ((r >> 2) << 4) + ((r & 3) << 1);
    union { unsigned short s[2]; unsigned u; } t;
    t.s[0] = *reinterpret_cast<const unsigned short*>(p + k * ld + col);
    t.s[1] = *reinterpret_cast<const unsigned short*>(p + (k + 1) * ld + col);
    f.u[r] = t.u;
  }
  return f.v;
}

// ---- fp32 -> bf16 weight pre-conversion ---------------------------------
__global__ void cvt_bf16_kernel(const float* __restrict__ in,
                                __bf16* __restrict__ out, int n) {
  const int i = blockIdx.x * 256 + threadIdx.x;
  if (i < n) out[i] = (__bf16)in[i];
}

// ---- LayerNorm: one wave per row of 384, float4 I/O, bf16 output --------
__global__ void __launch_bounds__(256) ln_kernel(const float* __restrict__ x,
                                                 const float* __restrict__ g,
                                                 const float* __restrict__ be,
                                                 __bf16* __restrict__ out) {
  const int row  = blockIdx.x * 8 + (threadIdx.x >> 5);
  const int lane = threadIdx.x & 31;
  const float4* xr4 = reinterpret_cast<const float4*>(x + (size_t)row * 384);
  const float4* g4  = reinterpret_cast<const float4*>(g);
  const float4* be4 = reinterpret_cast<const float4*>(be);

  float4 v[3];
  float sum = 0.f, sq = 0.f;
#pragma unroll
  for (int i = 0; i < 3; ++i) {
    float4 t = xr4[i * 32 + lane];
    v[i] = t;
    sum += t.x + t.y + t.z + t.w;
    sq  += t.x * t.x + t.y * t.y + t.z * t.z + t.w * t.w;
  }
#pragma unroll
  for (int off = 1; off < 32; off <<= 1) {
    sum += __shfl_xor(sum, off, 32);
    sq  += __shfl_xor(sq,  off, 32);
  }
  const float mean = sum * (1.f / 384.f);
  const float var  = sq * (1.f / 384.f) - mean * mean;
  const float rstd = rsqrtf(var + 1e-5f);

  uint2* orow = reinterpret_cast<uint2*>(out + (size_t)row * 384);
#pragma unroll
  for (int i = 0; i < 3; ++i) {
    const float4 gg = g4[i * 32 + lane];
    const float4 bb = be4[i * 32 + lane];
    union { __bf16 h[4]; uint2 u; } o;
    o.h[0] = (__bf16)((v[i].x - mean) * rstd * gg.x + bb.x);
    o.h[1] = (__bf16)((v[i].y - mean) * rstd * gg.y + bb.y);
    o.h[2] = (__bf16)((v[i].z - mean) * rstd * gg.z + bb.z);
    o.h[3] = (__bf16)((v[i].w - mean) * rstd * gg.w + bb.w);
    orow[i * 32 + lane] = o.u;
  }
}

// ---- Generic WMMA GEMM: D = act(A[MxK] * W[NxK]^T + bias) (+Res), bf16 in.
// Block 256 = 8 waves; wave -> 32x64 tile (2 A-frags, 4 B-frags, 8 WMMAs/k-step);
// block -> 256x64.
template <bool OUT_BF16, bool RELU, bool HAS_BIAS, bool HAS_RES>
__global__ void __launch_bounds__(256) gemm_wmma(const __bf16* __restrict__ A,
                                                 const __bf16* __restrict__ W,
                                                 const float* __restrict__ bias,
                                                 const float* __restrict__ Res,
                                                 void* __restrict__ Out,
                                                 int M, int N, int K) {
  const int wave = threadIdx.x >> 5;
  const int lane = threadIdx.x & 31;
  const int m0   = blockIdx.x * 256 + wave * 32;
  const int n0   = blockIdx.y * 64;

  v8f acc[2][4] = {};
  const __bf16* arow = A + (size_t)m0 * K;
  for (int k = 0; k < K; k += 32) {
    v16bf a0 = frag_row_bf16(arow + k, K);
    v16bf a1 = frag_row_bf16(arow + (size_t)16 * K + k, K);
#pragma unroll
    for (int j = 0; j < 4; ++j) {
      v16bf b = frag_row_bf16(W + (size_t)(n0 + 16 * j) * K + k, K);
      acc[0][j] = wmma_bf16(a0, b, acc[0][j]);
      acc[1][j] = wmma_bf16(a1, b, acc[1][j]);
    }
  }

  const int nloc = lane & 15;
  const int hi   = lane >> 4;
#pragma unroll
  for (int j = 0; j < 4; ++j) {
    const int n = n0 + 16 * j + nloc;
    const float bb = HAS_BIAS ? bias[n] : 0.f;
#pragma unroll
    for (int mi = 0; mi < 2; ++mi)
#pragma unroll
      for (int v = 0; v < 8; ++v) {
        const int m = m0 + 16 * mi + v + 8 * hi;
        float r = acc[mi][j][v] + bb;
        if (RELU) r = fmaxf(r, 0.f);
        if (HAS_RES) r += Res[(size_t)m * N + n];
        if (OUT_BF16) ((__bf16*)Out)[(size_t)m * N + n] = (__bf16)r;
        else          ((float*)Out)[(size_t)m * N + n] = r;
      }
  }
}

// ---- Causal attention: one wave per 16-query tile, 4 waves/block --------
// qkv: [B*T, 1152] bf16 (q|k|v, head h at col h*64).  out: [B*T, 384] bf16.
// qt is forced into an SGPR (readfirstlane) so causal tile skipping lowers
// to scalar s_cmp/s_cbranch with EXEC untouched (all-ones, as WMMA needs).
__global__ void __launch_bounds__(128, 1) attn_kernel(const __bf16* __restrict__ qkv,
                                                      __bf16* __restrict__ out) {
  __shared__ __bf16 Pbuf[4][16][256];   // 32 KB
  const int w    = threadIdx.x >> 5;
  const int lane = threadIdx.x & 31;
  const int b    = blockIdx.x / 6;
  const int hh   = blockIdx.x % 6;
  const int qt   = __builtin_amdgcn_readfirstlane(blockIdx.y * 4 + w); // wave-uniform
  const int q0   = qt * 16;
  const int qmax = q0 + 15;

  const __bf16* qbase = qkv + ((size_t)b * 256 + q0) * 1152 + hh * 64;
  const __bf16* kbase = qkv + ((size_t)b * 256) * 1152 + hh * 64 + 384;
  const __bf16* vbase = qkv + ((size_t)b * 256) * 1152 + hh * 64 + 768;

  // S = Q K^T  (16 x 256), K-dim = D = 64 -> 2 wmma steps per live 16-col tile
  v16bf aq0 = frag_row_bf16(qbase, 1152);
  v16bf aq1 = frag_row_bf16(qbase + 32, 1152);
  v8f s[16] = {};
#pragma unroll
  for (int jt = 0; jt < 16; ++jt) {
    if (jt <= qt) {                     // scalar branch: skip fully-masked tiles
      const __bf16* kp = kbase + (size_t)(jt * 16) * 1152;
      s[jt] = wmma_bf16(aq0, frag_row_bf16(kp, 1152), s[jt]);
      s[jt] = wmma_bf16(aq1, frag_row_bf16(kp + 32, 1152), s[jt]);
    }
  }

  // causal softmax; row m = v + 8*(lane>=16), col = 16*jt + (lane&15)
  const int nloc = lane & 15;
  const int hi   = lane >> 4;
#pragma unroll
  for (int v = 0; v < 8; ++v) {
    const int qrow = q0 + v + 8 * hi;
    float sv[16];
    float mx = -3.0e38f;
#pragma unroll
    for (int jt = 0; jt < 16; ++jt) {
      if (jt <= qt) {
        const int kcol = jt * 16 + nloc;
        float t = s[jt][v] * 0.125f;    // D^-0.5, D=64
        if (kcol > qrow) t = -3.0e38f;
        sv[jt] = t;
        mx = fmaxf(mx, t);
      } else {
        sv[jt] = -3.0e38f;
      }
    }
#pragma unroll
    for (int off = 1; off < 16; off <<= 1) mx = fmaxf(mx, __shfl_xor(mx, off, 32));
    float sum = 0.f;
#pragma unroll
    for (int jt = 0; jt < 16; ++jt) {
      if (jt <= qt) {
        float e = (sv[jt] < -1.0e37f) ? 0.f : __expf(sv[jt] - mx);
        sv[jt] = e; sum += e;
      } else {
        sv[jt] = 0.f;
      }
    }
#pragma unroll
    for (int off = 1; off < 16; off <<= 1) sum += __shfl_xor(sum, off, 32);
    const float inv = 1.f / sum;
#pragma unroll
    for (int jt = 0; jt < 16; ++jt) {
      if (jt <= (qt | 1))               // PV reads tile pairs: zero the sibling
        Pbuf[w][v + 8 * hi][jt * 16 + nloc] = (__bf16)(sv[jt] * inv);
    }
  }
  __syncthreads();

  // O = P V  (16 x 64), K-dim = 256 -> live 32-wide steps only
  v8f o[4] = {};
#pragma unroll
  for (int ksi = 0; ksi < 8; ++ksi) {
    const int ks = ksi * 32;
    if (ks <= qmax) {                   // scalar causal skip
      v16bf a = frag_row_bf16(&Pbuf[w][0][ks], 256);
#pragma unroll
      for (int dt = 0; dt < 4; ++dt) {
        v16bf bv = frag_col_bf16(vbase + (size_t)ks * 1152 + dt * 16, 1152);
        o[dt] = wmma_bf16(a, bv, o[dt]);
      }
    }
  }

#pragma unroll
  for (int dt = 0; dt < 4; ++dt)
#pragma unroll
    for (int v = 0; v < 8; ++v) {
      const int t = q0 + v + 8 * hi;
      out[((size_t)b * 256 + t) * 384 + hh * 64 + dt * 16 + nloc] = (__bf16)o[dt][v];
    }
}

extern "C" void kernel_launch(void* const* d_in, const int* in_sizes, int n_in,
                              void* d_out, int out_size, void* d_ws, size_t ws_size,
                              hipStream_t stream) {
  (void)in_sizes; (void)n_in; (void)out_size; (void)ws_size;
  const float* x     = (const float*)d_in[0];
  const float* Wqkv  = (const float*)d_in[1];
  const float* Wproj = (const float*)d_in[2];
  const float* bproj = (const float*)d_in[3];
  const float* W1    = (const float*)d_in[4];
  const float* b1    = (const float*)d_in[5];
  const float* W2    = (const float*)d_in[6];
  const float* b2    = (const float*)d_in[7];
  const float* g1    = (const float*)d_in[8];
  const float* be1   = (const float*)d_in[9];
  const float* g2    = (const float*)d_in[10];
  const float* be2   = (const float*)d_in[11];
  float* outp = (float*)d_out;

  const int M = 128 * 256;   // B*T = 32768 rows, C = 384

  // workspace layout (bf16 elems):
  //   h [M*384] | qkv [M*1152] | attout [M*384] | Wqkv_b | Wproj_b | W1_b | W2_b
  // ff1 [M*1536] reuses (qkv + attout) exactly.
  __bf16* h    = (__bf16*)d_ws;
  __bf16* qkv  = h + (size_t)M * 384;
  __bf16* ao   = qkv + (size_t)M * 1152;
  __bf16* ff1  = qkv;
  __bf16* Wq_b = ao + (size_t)M * 384;
  __bf16* Wp_b = Wq_b + 1152 * 384;
  __bf16* W1_b = Wp_b + 384 * 384;
  __bf16* W2_b = W1_b + 1536 * 384;

  dim3 blk256(256), blk128(128);

  // 0) pre-convert weights to bf16 (tiny: 1.77M elems total)
  cvt_bf16_kernel<<<(1152 * 384 + 255) / 256, blk256, 0, stream>>>(Wqkv, Wq_b, 1152 * 384);
  cvt_bf16_kernel<<<(384 * 384 + 255) / 256, blk256, 0, stream>>>(Wproj, Wp_b, 384 * 384);
  cvt_bf16_kernel<<<(1536 * 384 + 255) / 256, blk256, 0, stream>>>(W1, W1_b, 1536 * 384);
  cvt_bf16_kernel<<<(384 * 1536 + 255) / 256, blk256, 0, stream>>>(W2, W2_b, 384 * 1536);

  // 1) h = LN(x)
  ln_kernel<<<M / 8, blk256, 0, stream>>>(x, g1, be1, h);
  // 2) qkv = h @ Wqkv^T           [M x 1152]
  gemm_wmma<true, false, false, false><<<dim3(M / 256, 1152 / 64), blk256, 0, stream>>>(
      h, Wq_b, nullptr, nullptr, qkv, M, 1152, 384);
  // 3) attention -> ao            [M x 384]
  attn_kernel<<<dim3(128 * 6, 4), blk128, 0, stream>>>(qkv, ao);
  // 4) x2 = x + ao @ Wproj^T + bproj  -> d_out (fp32)
  gemm_wmma<false, false, true, true><<<dim3(M / 256, 384 / 64), blk256, 0, stream>>>(
      ao, Wp_b, bproj, x, outp, M, 384, 384);
  // 5) h = LN(x2)
  ln_kernel<<<M / 8, blk256, 0, stream>>>(outp, g2, be2, h);
  // 6) ff1 = relu(h @ W1^T + b1)  [M x 1536]
  gemm_wmma<true, true, true, false><<<dim3(M / 256, 1536 / 64), blk256, 0, stream>>>(
      h, W1_b, b1, nullptr, ff1, M, 1536, 384);
  // 7) out = x2 + ff1 @ W2^T + b2 -> d_out (fp32)
  gemm_wmma<false, false, true, true><<<dim3(M / 256, 384 / 64), blk256, 0, stream>>>(
      ff1, W2_b, b2, outp, outp, M, 384, 1536);
}
